// DenseGeneral_87703232185040
// MI455X (gfx1250) — 8-bit matrix operand order repaired and checked by reading, compiled, not run on hardware
//
#include <hip/hip_runtime.h>
#include <stdint.h>

#define M_DIM 8192
#define K_DIM 2048
#define N_DIM 8192

typedef __attribute__((ext_vector_type(16))) int   v16i;
typedef __attribute__((ext_vector_type(8)))  float v8f;

#if defined(__gfx1250__) && __has_builtin(__builtin_amdgcn_global_load_async_to_lds_b128) && \
    __has_builtin(__builtin_amdgcn_s_wait_asynccnt)
#define USE_ASYNC_LDS 1
#else
#define USE_ASYNC_LDS 0
#endif

typedef int v4i_vs __attribute__((vector_size(16)));
typedef __attribute__((address_space(1))) v4i_vs v4i_glb;
typedef __attribute__((address_space(3))) v4i_vs v4i_lds;

__device__ __forceinline__ void copy16_g2l(const uint8_t* g, uint8_t* l) {
#if USE_ASYNC_LDS
  __builtin_amdgcn_global_load_async_to_lds_b128(
      (v4i_glb*)(unsigned long long)(uintptr_t)g,
      (v4i_lds*)(unsigned int)(uintptr_t)l,
       0,  0);
#else
  *(uint4*)l = *(const uint4*)g;
#endif
}

__device__ __forceinline__ float clamp448(float v) {
  return fminf(fmaxf(v, -448.0f), 448.0f);
}

__device__ __forceinline__ unsigned e4m3_byte(float x) {
  unsigned u = __float_as_uint(x);
  unsigned s = (u >> 24) & 0x80u;
  float ax = fabsf(x);
  if (ax < 0.015625f) {
    return s | (unsigned)rintf(ax * 512.0f);
  }
  unsigned au = u & 0x7FFFFFFFu;
  au += 0x7FFFFu + ((au >> 20) & 1u);
  return s | (((au >> 20) - 960u) & 0x7Fu);
}

__device__ __forceinline__ unsigned pack4_e4m3(float a, float b, float c, float d, float inv) {
  a = clamp448(a * inv); b = clamp448(b * inv);
  c = clamp448(c * inv); d = clamp448(d * inv);
#if __has_builtin(__builtin_amdgcn_cvt_pk_fp8_f32)
  int lo = __builtin_amdgcn_cvt_pk_fp8_f32(a, b, 0, false);
  int v  = __builtin_amdgcn_cvt_pk_fp8_f32(c, d, lo, true);
  return (unsigned)v;
#else
  return e4m3_byte(a) | (e4m3_byte(b) << 8) | (e4m3_byte(c) << 16) | (e4m3_byte(d) << 24);
#endif
}

__device__ __forceinline__ float bf16_roundtrip(float b) {
  unsigned u = __float_as_uint(b);
  unsigned r = (u + 0x7FFFu + ((u >> 16) & 1u)) & 0xFFFF0000u;
  return __uint_as_float(r);
}

__global__ void __launch_bounds__(256) quant_e4m3_kernel(
    const float* __restrict__ x, uint8_t* __restrict__ q,
    const float* __restrict__ scale, int n16) {
  const float inv = 1.0f / scale[0];
  int i = blockIdx.x * 256 + threadIdx.x;
  if (i >= n16) return;
  const float4* xp = (const float4*)x + (size_t)i * 4;
  float4 f0 = xp[0], f1 = xp[1], f2 = xp[2], f3 = xp[3];
  uint4 o;
  o.x = pack4_e4m3(f0.x, f0.y, f0.z, f0.w, inv);
  o.y = pack4_e4m3(f1.x, f1.y, f1.z, f1.w, inv);
  o.z = pack4_e4m3(f2.x, f2.y, f2.z, f2.w, inv);
  o.w = pack4_e4m3(f3.x, f3.y, f3.z, f3.w, inv);
  ((uint4*)q)[i] = o;
}

__global__ void __launch_bounds__(256) quantT_e4m3_kernel(
    const float* __restrict__ x,
    uint8_t* __restrict__ q,
    const float* __restrict__ scale) {
  __shared__ __align__(16) uint8_t tile[64][68];
  const float inv = 1.0f / scale[0];
  const int f0 = blockIdx.x * 64;
  const int k0 = blockIdx.y * 64;
  const int t  = threadIdx.x;
  const int cg = t & 15;
  const int r0 = t >> 4;
#pragma unroll
  for (int i = 0; i < 4; ++i) {
    int r = r0 + i * 16;
    float4 f = *(const float4*)(x + (size_t)(k0 + r) * N_DIM + f0 + cg * 4);
    *(unsigned*)&tile[r][cg * 4] = pack4_e4m3(f.x, f.y, f.z, f.w, inv);
  }
  __syncthreads();
  const int n  = t & 63;
  const int kg = t >> 6;
  unsigned b[16];
#pragma unroll
  for (int j = 0; j < 16; ++j) b[j] = tile[kg * 16 + j][n];
  uint4 o;
  o.x = b[0]  | (b[1] << 8)  | (b[2] << 16)  | (b[3] << 24);
  o.y = b[4]  | (b[5] << 8)  | (b[6] << 16)  | (b[7] << 24);
  o.z = b[8]  | (b[9] << 8)  | (b[10] << 16) | (b[11] << 24);
  o.w = b[12] | (b[13] << 8) | (b[14] << 16) | (b[15] << 24);
  *(uint4*)(q + (size_t)(f0 + n) * K_DIM + k0 + kg * 16) = o;
}

#define BM 128
#define BN 256
#define BK 128
#define NKT (K_DIM / BK)

__global__ void __launch_bounds__(256, 1) gemm_fp8_kernel(
    const uint8_t* __restrict__ A,
    const uint8_t* __restrict__ Bt,
    float* __restrict__ out,
    const float* __restrict__ bias,
    const float* __restrict__ sA, const float* __restrict__ sB) {
  __shared__ __align__(16) uint8_t Als[2][BM * BK];
  __shared__ __align__(16) uint8_t Bls[2][BN * BK];

  const int tid   = threadIdx.x;
  const int wave  = tid >> 5;
  const int lane  = tid & 31;
  const int ln    = lane & 15;
  const int hi    = lane >> 4;
  const int waveM = wave >> 2;
  const int waveN = wave & 3;
  const int m0    = blockIdx.y * BM;
  const int n0    = blockIdx.x * BN;

  auto stage = [&](int buf, int k0) {
#pragma unroll
    for (int i = 0; i < 4; ++i) {
      int qi = tid + i * 256;
      int r = qi >> 3, kc = qi & 7;
      copy16_g2l(A + (size_t)(m0 + r) * K_DIM + k0 + kc * 16,
                 &Als[buf][r * BK + kc * 16]);
    }
#pragma unroll
    for (int i = 0; i < 8; ++i) {
      int qi = tid + i * 256;
      int r = qi >> 3, kc = qi & 7;
      copy16_g2l(Bt + (size_t)(n0 + r) * K_DIM + k0 + kc * 16,
                 &Bls[buf][r * BK + kc * 16]);
    }
  };

  v8f acc[4][4];
#pragma unroll
  for (int i = 0; i < 4; ++i)
#pragma unroll
    for (int j = 0; j < 4; ++j) acc[i][j] = (v8f)(0.0f);

  stage(0, 0);

#pragma unroll 1
  for (int kt = 0; kt < NKT; ++kt) {
    const int cur = kt & 1;

    if (kt + 1 < NKT) {
      stage(cur ^ 1, (kt + 1) * BK);
#if USE_ASYNC_LDS
      __builtin_amdgcn_s_wait_asynccnt(12);
#endif
    } else {
#if USE_ASYNC_LDS
      __builtin_amdgcn_s_wait_asynccnt(0);
#endif
    }
    __syncthreads();

    v16i af[4];
#pragma unroll
    for (int t = 0; t < 4; ++t) {
      const uint8_t* base = &Als[cur][(size_t)(waveM * 64 + t * 16 + ln) * BK];
#pragma unroll
      for (int c = 0; c < 8; ++c) {
        uint2 d = *(const uint2*)(base + (c >> 1) * 32 + hi * 16 + (c & 1) * 8);
        af[t][2 * c]     = (int)d.x;
        af[t][2 * c + 1] = (int)d.y;
      }
    }
    v16i bf[4];
#pragma unroll
    for (int t = 0; t < 4; ++t) {
      const uint8_t* base = &Bls[cur][(size_t)(waveN * 64 + t * 16 + ln) * BK];
#pragma unroll
      for (int c = 0; c < 4; ++c) {
        uint4 d = *(const uint4*)(base + c * 32 + hi * 16);
        bf[t][4 * c]     = (int)d.x;
        bf[t][4 * c + 1] = (int)d.y;
        bf[t][4 * c + 2] = (int)d.z;
        bf[t][4 * c + 3] = (int)d.w;
      }
    }

#pragma unroll
    for (int ti = 0; ti < 4; ++ti)
#pragma unroll
      for (int tj = 0; tj < 4; ++tj)
        acc[ti][tj] = __builtin_amdgcn_wmma_f32_16x16x128_fp8_fp8(
            af[ti], bf[tj], (short)0, acc[ti][tj],
             false,  false);

    __syncthreads();
  }

  const float s = sA[0] * sB[0];
#pragma unroll
  for (int tj = 0; tj < 4; ++tj) {
    const int n = n0 + waveN * 64 + tj * 16 + ln;
    const float bb = bf16_roundtrip(bias[n]);
#pragma unroll
    for (int ti = 0; ti < 4; ++ti) {
      const int mbase = m0 + waveM * 64 + ti * 16 + hi * 8;
#pragma unroll
      for (int r = 0; r < 8; ++r) {
        out[(size_t)(mbase + r) * N_DIM + n] = acc[ti][tj][r] * s + bb;
      }
    }
  }
}

extern "C" void kernel_launch(void* const* d_in, const int* in_sizes, int n_in,
                              void* d_out, int out_size, void* d_ws, size_t ws_size,
                              hipStream_t stream) {
  const float* inputs   = (const float*)d_in[0];
  const float* kern     = (const float*)d_in[1];
  const float* bias     = (const float*)d_in[2];
  const float* in_scale = (const float*)d_in[3];
  const float* k_scale  = (const float*)d_in[4];

  uint8_t* wsA = (uint8_t*)d_ws;
  uint8_t* wsB = wsA + (size_t)M_DIM * K_DIM;

  {
    int n16 = (M_DIM * K_DIM) / 16;
    quant_e4m3_kernel<<<n16 / 256, 256, 0, stream>>>(inputs, wsA, in_scale, n16);
  }
  {
    dim3 grid(N_DIM / 64, K_DIM / 64);
    quantT_e4m3_kernel<<<grid, 256, 0, stream>>>(kern, wsB, k_scale);
  }
  {
    dim3 grid(N_DIM / BN, M_DIM / BM);
    gemm_fp8_kernel<<<grid, 256, 0, stream>>>(wsA, wsB, (float*)d_out, bias,
                                              in_scale, k_scale);
  }
}
